// AttentionMech_27968827031773
// MI455X (gfx1250) — matrix operand order repaired and checked by reading, not compiled, not run on hardware
//
#include <hip/hip_runtime.h>
#include <math.h>

// ---------------- problem constants ----------------
#define BB 4
#define CC 256
#define ACC 64
#define NN 4096

typedef __bf16 bf16_t;
typedef __attribute__((ext_vector_type(16))) __bf16 v16bf;
typedef __attribute__((ext_vector_type(8)))  float  v8f;

// ---------------- workspace layout (bytes) ----------------
#define XBT_OFF  0u                       // [B][N][C] bf16 : 8,388,608
#define QT_OFF   8388608u                 // [B][N][AC] bf16: 2,097,152
#define KT_OFF   10485760u                // [B][N][AC] bf16: 2,097,152
#define VV_OFF   12582912u                // [B][C][N] bf16 : 8,388,608
#define WQB_OFF  20971520u                // [AC][C] bf16   : 32,768
#define WKB_OFF  21004288u
#define WVB_OFF  21037056u                // [C][C] bf16    : 131,072
#define CMAX_OFF 21168128u                // [B][N] f32     : 65,536
#define CINV_OFF 21233664u                // [B][N] f32 (1/sumexp): 65,536

// ---------------- WMMA helpers ----------------
__device__ __forceinline__ v8f wmma_bf16(v16bf a, v16bf b, v8f c) {
  // D = A(16x32 bf16) * B(32x16 bf16) + C(16x16 f32)
  return __builtin_amdgcn_wmma_f32_16x16x32_bf16(false, a, false, b, (short)0, c,
                                                 false, false);
}

// A fragment: 16x32 (MxK) bf16, source row-major [M][ld], element (m,k)=src[m*ld+k].
// ISA 16-bit A table: lanes 0-15: M=lane, vgpr0-3 K=0..7, vgpr4-7 K=16..23;
// lanes 16-31: M=lane-16, vgpr0-3 K=8..15, vgpr4-7 K=24..31. Pairs are K-contiguous.
__device__ __forceinline__ v16bf load_frag_a(const bf16_t* src, int ld,
                                             int mBase, int kBase, int lane) {
  int m = mBase + (lane & 15);
  int kHi = (lane >> 4) * 8;
  v16bf f;
#pragma unroll
  for (int v = 0; v < 8; ++v) {
    int k = kBase + kHi + ((v >= 4) ? 16 : 0) + 2 * (v & 3);
    f[2 * v]     = src[(size_t)m * ld + k];
    f[2 * v + 1] = src[(size_t)m * ld + k + 1];
  }
  return f;
}

// B fragment: 32x16 (KxN) bf16, loaded from memory organized as Bmem[N][Kdim]
// (B transposed, K contiguous). Same K order as the A table:
// lanes 0-15: N=lane, K=0..7,16..23; lanes 16-31: N=lane-16, K=8..15,24..31.
__device__ __forceinline__ v16bf load_frag_b(const bf16_t* bmem, int ld,
                                             int kBase, int nBase, int lane) {
  int n  = nBase + (lane & 15);
  int k0 = kBase + (lane >> 4) * 8;
  v16bf f;
#pragma unroll
  for (int h = 0; h < 16; ++h) f[h] = bmem[(size_t)n * ld + k0 + h + ((h >= 8) ? 8 : 0)];
  return f;
}

__device__ __forceinline__ v8f zero8() {
  v8f z = {0.f, 0.f, 0.f, 0.f, 0.f, 0.f, 0.f, 0.f};
  return z;
}

// ---------------- conversion kernels ----------------
__global__ void convert_f32_bf16(const float* __restrict__ src,
                                 bf16_t* __restrict__ dst, int count) {
  int i = blockIdx.x * blockDim.x + threadIdx.x;
  if (i < count) dst[i] = (bf16_t)src[i];
}

// x[b][c][n] f32  ->  xbT[b][n][c] bf16 (coalesced reads along n)
__global__ void transpose_convert_x(const float* __restrict__ x,
                                    bf16_t* __restrict__ xbT) {
  int idx = blockIdx.x * blockDim.x + threadIdx.x;   // b*C*N total
  int n = idx % NN;
  int c = (idx / NN) % CC;
  int b = idx / (NN * CC);
  xbT[((size_t)b * NN + n) * CC + c] = (bf16_t)x[idx];
}

// ---------------- QKV GEMMs ----------------
// outT[b][n][o] = sum_c xbT[b][n][c] * W[o][c] + bias[o]   (o-dim = ACC)
__global__ __launch_bounds__(32) void qk_gemm_kernel(const bf16_t* __restrict__ xbT,
                                                     const bf16_t* __restrict__ wb,
                                                     const float* __restrict__ bias,
                                                     bf16_t* __restrict__ outT) {
  int lane = threadIdx.x;
  int nBase = blockIdx.x * 16;
  int oBase = blockIdx.y * 16;
  int b = blockIdx.z;
  const bf16_t* xb = xbT + (size_t)b * NN * CC;
  v8f acc = zero8();
#pragma unroll 4
  for (int kb = 0; kb < CC; kb += 32) {
    v16bf a  = load_frag_a(xb, CC, nBase, kb, lane);   // A[n,c]
    v16bf bw = load_frag_b(wb, CC, kb, oBase, lane);   // B[c,o] from W[o][c]
    acc = wmma_bf16(a, bw, acc);
  }
  int col = lane & 15, rowOff = (lane >> 4) * 8;
  float bs = bias[oBase + col];
  bf16_t* ob = outT + (size_t)b * NN * ACC;
#pragma unroll
  for (int r = 0; r < 8; ++r)
    ob[(size_t)(nBase + rowOff + r) * ACC + oBase + col] = (bf16_t)(acc[r] + bs);
}

// v[b][o][n] = sum_c Wv[o][c] * xbT[b][n][c] + bv[o]   (o-dim = CC)
__global__ __launch_bounds__(32) void v_gemm_kernel(const bf16_t* __restrict__ xbT,
                                                    const bf16_t* __restrict__ wvb,
                                                    const float* __restrict__ bv,
                                                    bf16_t* __restrict__ vout) {
  int lane = threadIdx.x;
  int nBase = blockIdx.x * 16;
  int oBase = blockIdx.y * 16;
  int b = blockIdx.z;
  const bf16_t* xb = xbT + (size_t)b * NN * CC;
  v8f acc = zero8();
#pragma unroll 4
  for (int kb = 0; kb < CC; kb += 32) {
    v16bf a  = load_frag_a(wvb, CC, oBase, kb, lane);  // A[o,c]
    v16bf bx = load_frag_b(xb, CC, kb, nBase, lane);   // B[c,n] from xbT[n][c]
    acc = wmma_bf16(a, bx, acc);
  }
  int col = lane & 15, rowOff = (lane >> 4) * 8;
  bf16_t* vb = vout + (size_t)b * CC * NN;
#pragma unroll
  for (int r = 0; r < 8; ++r) {
    int o = oBase + rowOff + r;
    vb[(size_t)o * NN + nBase + col] = (bf16_t)(acc[r] + bv[o]);
  }
}

// ---------------- column softmax stats: max + 1/sumexp over i, per column j ----------
__global__ __launch_bounds__(32) void stats_kernel(const bf16_t* __restrict__ qT,
                                                   const bf16_t* __restrict__ kT,
                                                   float* __restrict__ cmax,
                                                   float* __restrict__ cinv) {
  int lane = threadIdx.x;
  int jBase = blockIdx.x * 16;
  int b = blockIdx.z;
  const bf16_t* qTb = qT + (size_t)b * NN * ACC;
  const bf16_t* kTb = kT + (size_t)b * NN * ACC;
  // B fragments (K=64 -> two) for this column tile, invariant over i loop
  v16bf bK0 = load_frag_b(kTb, ACC, 0,  jBase, lane);
  v16bf bK1 = load_frag_b(kTb, ACC, 32, jBase, lane);
  float m = -__builtin_inff();
  float l = 0.f;
  for (int ib = 0; ib < NN; ib += 16) {
    v16bf a0 = load_frag_a(qTb, ACC, ib, 0,  lane);
    v16bf a1 = load_frag_a(qTb, ACC, ib, 32, lane);
    v8f s = zero8();
    s = wmma_bf16(a0, bK0, s);
    s = wmma_bf16(a1, bK1, s);
    float tmax = s[0];
#pragma unroll
    for (int r = 1; r < 8; ++r) tmax = fmaxf(tmax, s[r]);
    float nm = fmaxf(m, tmax);
    float add = 0.f;
#pragma unroll
    for (int r = 0; r < 8; ++r) add += __expf(s[r] - nm);
    l = l * __expf(m - nm) + add;
    m = nm;
  }
  // combine the two lanes (lane, lane^16) holding the same column
  float mo = __shfl_xor(m, 16, 32);
  float lo = __shfl_xor(l, 16, 32);
  float nm = fmaxf(m, mo);
  l = l * __expf(m - nm) + lo * __expf(mo - nm);
  if (lane < 16) {
    cmax[(size_t)b * NN + jBase + lane] = nm;
    cinv[(size_t)b * NN + jBase + lane] = 1.f / l;
  }
}

// ---------------- PV: O[i,c] = sum_j softmax(S)[i,j] * v[c,j]; out = gamma*O + x ----------
// One wave = 16 output rows x 128 output channels (8 f32 accumulators).
__global__ __launch_bounds__(32) void pv_kernel(const bf16_t* __restrict__ qT,
                                                const bf16_t* __restrict__ kT,
                                                const bf16_t* __restrict__ vmat,
                                                const float* __restrict__ cmax,
                                                const float* __restrict__ cinv,
                                                const float* __restrict__ x,
                                                const float* __restrict__ gamma_p,
                                                float* __restrict__ out) {
  int lane = threadIdx.x;
  int iBase = blockIdx.x * 16;    // 16 output rows (spatial index i)
  int cHalf = blockIdx.y;         // 128 output channels per block
  int b = blockIdx.z;
  const bf16_t* qTb = qT + (size_t)b * NN * ACC;
  const bf16_t* kTb = kT + (size_t)b * NN * ACC;
  const bf16_t* vb  = vmat + (size_t)b * CC * NN;
  const float* mb  = cmax + (size_t)b * NN;
  const float* ilb = cinv + (size_t)b * NN;

  __shared__ bf16_t Plds[16 * 32];   // P tile, row-major [i][j], 1 KB

  // q A-fragments for this row tile (K=64), loop-invariant
  v16bf aQ0 = load_frag_a(qTb, ACC, iBase, 0,  lane);
  v16bf aQ1 = load_frag_a(qTb, ACC, iBase, 32, lane);

  v8f accO[8];
#pragma unroll
  for (int t = 0; t < 8; ++t) accO[t] = zero8();

  int col = lane & 15, rowOff = (lane >> 4) * 8;
  int cBase0 = cHalf * 128;

#pragma unroll 1
  for (int jb = 0; jb < NN; jb += 32) {
    // prefetch next j-chunk of kT (32 rows x 64 ck bf16 = 4 KB, 32 lanes x 128 B)
    if (jb + 32 < NN) {
      const char* pf = (const char*)(kTb + (size_t)(jb + 32) * ACC) + lane * 128;
      __builtin_prefetch(pf, 0, 1);
    }

    // ---- S tile 16x32 (two 16x16 WMMA accumulators), K = 64 ----
    v16bf bK00 = load_frag_b(kTb, ACC, 0,  jb,      lane);
    v16bf bK01 = load_frag_b(kTb, ACC, 32, jb,      lane);
    v16bf bK10 = load_frag_b(kTb, ACC, 0,  jb + 16, lane);
    v16bf bK11 = load_frag_b(kTb, ACC, 32, jb + 16, lane);
    v8f s0 = zero8(), s1 = zero8();
    s0 = wmma_bf16(aQ0, bK00, s0);
    s0 = wmma_bf16(aQ1, bK01, s0);
    s1 = wmma_bf16(aQ0, bK10, s1);
    s1 = wmma_bf16(aQ1, bK11, s1);

    // ---- normalize with precomputed column stats; P -> LDS (relayout C->A frag) ----
    float m0  = mb[jb + col];
    float il0 = ilb[jb + col];
    float m1  = mb[jb + 16 + col];
    float il1 = ilb[jb + 16 + col];
    __syncthreads();
#pragma unroll
    for (int r = 0; r < 8; ++r) {
      int i = rowOff + r;
      Plds[i * 32 + col]      = (bf16_t)(__expf(s0[r] - m0) * il0);
      Plds[i * 32 + 16 + col] = (bf16_t)(__expf(s1[r] - m1) * il1);
    }
    __syncthreads();

    // ---- O += P(16x32) * v^T(32x16 per c-tile), 8 c-tiles ----
    v16bf aP = load_frag_a(Plds, 32, 0, 0, lane);
#pragma unroll
    for (int t = 0; t < 8; ++t) {
      int cBase = cBase0 + t * 16;
      v16bf bV = load_frag_b(vb, NN, jb, cBase, lane);  // B[j,c] from v[c][j]
      accO[t] = wmma_bf16(aP, bV, accO[t]);
    }
  }

  // ---- epilogue: out[b][c][i] = gamma * O[i][c] + x[b][c][i] ----
  float g = gamma_p[0];
#pragma unroll
  for (int t = 0; t < 8; ++t) {
    int cGlob = cBase0 + t * 16 + col;
#pragma unroll
    for (int r = 0; r < 8; ++r) {
      int i = iBase + rowOff + r;
      size_t idx = ((size_t)b * CC + cGlob) * NN + i;
      out[idx] = g * accO[t][r] + x[idx];
    }
  }
}

// ---------------- launcher ----------------
extern "C" void kernel_launch(void* const* d_in, const int* in_sizes, int n_in,
                              void* d_out, int out_size, void* d_ws, size_t ws_size,
                              hipStream_t stream) {
  const float* x     = (const float*)d_in[0];
  const float* Wq    = (const float*)d_in[1];
  const float* bq    = (const float*)d_in[2];
  const float* Wk    = (const float*)d_in[3];
  const float* bk    = (const float*)d_in[4];
  const float* Wv    = (const float*)d_in[5];
  const float* bv    = (const float*)d_in[6];
  const float* gamma = (const float*)d_in[7];
  float* out = (float*)d_out;

  char* ws = (char*)d_ws;
  bf16_t* xbT  = (bf16_t*)(ws + XBT_OFF);
  bf16_t* qT   = (bf16_t*)(ws + QT_OFF);
  bf16_t* kT   = (bf16_t*)(ws + KT_OFF);
  bf16_t* vmat = (bf16_t*)(ws + VV_OFF);
  bf16_t* wqb  = (bf16_t*)(ws + WQB_OFF);
  bf16_t* wkb  = (bf16_t*)(ws + WKB_OFF);
  bf16_t* wvb  = (bf16_t*)(ws + WVB_OFF);
  float*  cmax = (float*)(ws + CMAX_OFF);
  float*  cinv = (float*)(ws + CINV_OFF);

  // 1) precision conversion / packing
  transpose_convert_x<<<(BB * CC * NN) / 256, 256, 0, stream>>>(x, xbT);
  convert_f32_bf16<<<(ACC * CC + 255) / 256, 256, 0, stream>>>(Wq, wqb, ACC * CC);
  convert_f32_bf16<<<(ACC * CC + 255) / 256, 256, 0, stream>>>(Wk, wkb, ACC * CC);
  convert_f32_bf16<<<(CC * CC + 255) / 256, 256, 0, stream>>>(Wv, wvb, CC * CC);

  // 2) QKV channel GEMMs (WMMA)
  qk_gemm_kernel<<<dim3(NN / 16, ACC / 16, BB), 32, 0, stream>>>(xbT, wqb, bq, qT);
  qk_gemm_kernel<<<dim3(NN / 16, ACC / 16, BB), 32, 0, stream>>>(xbT, wkb, bk, kT);
  v_gemm_kernel<<<dim3(NN / 16, CC / 16, BB), 32, 0, stream>>>(xbT, wvb, bv, vmat);

  // 3) column softmax statistics (max + 1/sumexp over i, per column j)
  stats_kernel<<<dim3(NN / 16, 1, BB), 32, 0, stream>>>(qT, kT, cmax, cinv);

  // 4) fused recompute-S + normalize + P*V^T + gamma*O + x
  pv_kernel<<<dim3(NN / 16, 2, BB), 32, 0, stream>>>(qT, kT, vmat, cmax, cinv,
                                                     x, gamma, out);
}
